// DKAModule_1030792151379
// MI455X (gfx1250) — matrix operand order repaired and checked by reading, not compiled, not run on hardware
//
#include <hip/hip_runtime.h>
#include <stdint.h>

typedef __attribute__((ext_vector_type(16))) _Float16 v16h;
typedef __attribute__((ext_vector_type(8)))  _Float16 v8h;
typedef __attribute__((ext_vector_type(4)))  _Float16 v4h;
typedef __attribute__((ext_vector_type(8)))  float    v8f;
typedef __attribute__((ext_vector_type(4)))  float    v4f;
typedef __attribute__((ext_vector_type(4)))  int      v4i;

#define D_MODEL 1024
#define NHEADS  8
#define DHEAD   128
#define RANK    4
#define BATCH   4
#define SEQ     4096
#define NTOK    (BATCH * SEQ)

// -------- CDNA5 async global->LDS copy (ASYNCcnt path), guarded ------------
#if defined(__has_builtin)
#if __has_builtin(__builtin_amdgcn_global_load_async_to_lds_b128)
#define HAVE_ASYNC_LDS 1
#endif
#endif

__device__ inline void async_b128(const _Float16* g, _Float16* l) {
#if HAVE_ASYNC_LDS
  __builtin_amdgcn_global_load_async_to_lds_b128(
      (__attribute__((address_space(1))) v4i*)(uintptr_t)g,
      (__attribute__((address_space(3))) v4i*)(uint32_t)(uintptr_t)l, 0, 0);
#endif
}

__device__ inline void wait_async0() {
#if defined(__HIP_DEVICE_COMPILE__)
#if defined(__has_builtin) && __has_builtin(__builtin_amdgcn_s_wait_asynccnt)
  __builtin_amdgcn_s_wait_asynccnt(0);
#else
  asm volatile("s_wait_asynccnt 0" ::: "memory");
#endif
#endif
}

// ---------------------------------------------------------------------------
// f32 -> f16 conversion, 4 elements per thread
// ---------------------------------------------------------------------------
__global__ void cvt_f32_f16(const float* __restrict__ in,
                            _Float16* __restrict__ out, int n4) {
  int i = blockIdx.x * blockDim.x + threadIdx.x;
  if (i < n4) {
    v4f v = *(const v4f*)(in + (size_t)i * 4);
    v4h h = {(_Float16)v.x, (_Float16)v.y, (_Float16)v.z, (_Float16)v.w};
    *(v4h*)(out + (size_t)i * 4) = h;
  }
}

// ---------------------------------------------------------------------------
// out[n][k] = (f16) in[k][n]  (square dim x dim)
// ---------------------------------------------------------------------------
__global__ void transpose_cvt(const float* __restrict__ in,
                              _Float16* __restrict__ out, int dim) {
  int k = blockIdx.x * blockDim.x + threadIdx.x;
  int n = blockIdx.y;
  if (k < dim)
    out[(size_t)n * dim + k] = (_Float16)in[(size_t)k * dim + n];
}

// ---------------------------------------------------------------------------
// C[M,N] = A[M,K] f16 row-major x BT[N,K] (B transposed) + bias, fp32 out.
// Block = 8 waves = 256x64 output tile. B slab (64 cols x 32 k, 4KB) is
// double-buffered in LDS via GLOBAL_LOAD_ASYNC_TO_LDS_B128 (fallback: sync
// staging). Each wave: 32x64 register tile, 8 WMMAs per k-step of 32.
// A frag (16-bit 16x32): lane m+16g holds row m, halfs [0..15]=K(g*16..+15).
//   B frag (32x16): lane n+16g holds col n, K=g*16..+15.
// ---------------------------------------------------------------------------
__global__ void __launch_bounds__(256) gemm_f16_wmma(
    const _Float16* __restrict__ A, const _Float16* __restrict__ BT,
    const float* __restrict__ bias, float* __restrict__ C,
    int M, int Nn, int K) {
  __shared__ __align__(32) _Float16 sB[2][64][32];

  int tid  = threadIdx.x;
  int wv   = tid >> 5;
  int lane = tid & 31;
  int g    = lane >> 4;
  int l    = lane & 15;

  int ntg   = Nn >> 6;                 // 64-column groups
  int mt    = blockIdx.x / ntg;
  int ng    = blockIdx.x - mt * ntg;
  int mbase = mt * 256 + wv * 32;
  int n0    = ng * 64;

  const _Float16* arow0 = A + (size_t)(mbase + l) * K;
  const _Float16* arow1 = A + (size_t)(mbase + 16 + l) * K;

  // this thread's B-staging slot: one b128 (8 halfs) per k-step
  int bcol  = tid >> 2;
  int bpart = tid & 3;
  const _Float16* bsrc = BT + (size_t)(n0 + bcol) * K + bpart * 8;
  _Float16* bdst0 = &sB[0][bcol][bpart * 8];
  _Float16* bdst1 = &sB[1][bcol][bpart * 8];

  v8f acc[2][4];
#pragma unroll
  for (int h = 0; h < 2; h++)
#pragma unroll
    for (int t = 0; t < 4; t++) acc[h][t] = (v8f){};

#if HAVE_ASYNC_LDS
  async_b128(bsrc, bdst0);
  wait_async0();
  __syncthreads();
#endif

  for (int k0 = 0; k0 < K; k0 += 32) {
    int buf = (k0 >> 5) & 1;
#if HAVE_ASYNC_LDS
    if (k0 + 32 < K) async_b128(bsrc + k0 + 32, buf ? bdst0 : bdst1);
#else
    buf = 0;
    *(v8h*)bdst0 = *(const v8h*)(bsrc + k0);
    __syncthreads();
#endif

    v16h a0  = *(const v16h*)(arow0 + k0 + g * 16);
    v16h a1  = *(const v16h*)(arow1 + k0 + g * 16);

#pragma unroll
    for (int t = 0; t < 4; t++) {
      v16h b = *(const v16h*)(&sB[buf][t * 16 + l][g * 16]);
      acc[0][t] = __builtin_amdgcn_wmma_f32_16x16x32_f16(
          false, a0, false, b, (short)0, acc[0][t], false, false);
      acc[1][t] = __builtin_amdgcn_wmma_f32_16x16x32_f16(
          false, a1, false, b, (short)0, acc[1][t], false, false);
    }

#if HAVE_ASYNC_LDS
    wait_async0();   // prefetch for next buffer has landed
    __syncthreads(); // everyone done reading current buffer
#else
    __syncthreads();
#endif
  }

  float bv0 = bias[n0 + l];
  float bv1 = bias[n0 + l + 16];
  float bv2 = bias[n0 + l + 32];
  float bv3 = bias[n0 + l + 48];
#pragma unroll
  for (int h = 0; h < 2; h++) {
#pragma unroll
    for (int r = 0; r < 8; r++) {
      float* cr = C + (size_t)(mbase + h * 16 + r + g * 8) * Nn + n0 + l;
      cr[0]  = acc[h][0][r] + bv0;
      cr[16] = acc[h][1][r] + bv1;
      cr[32] = acc[h][2][r] + bv2;
      cr[48] = acc[h][3][r] + bv3;
    }
  }
}

// ---------------------------------------------------------------------------
// Per-head dynamic-kernel mixing. One wave per token; each lane owns 4
// channels (d0 = lane*4). Templated on tap count KS so logits live in regs.
// ---------------------------------------------------------------------------
template <int KS>
__global__ void __launch_bounds__(256) dka_head(
    const float* __restrict__ xproj, const float* __restrict__ W,
    const float* __restrict__ bvec, const float* __restrict__ V,
    const float* __restrict__ S, const float* __restrict__ alpha_p,
    _Float16* __restrict__ mixed, int head) {
  constexpr int KR  = KS * RANK;
  constexpr int PAD = KS / 2;

  __shared__ float s_kst[KS * DHEAD];
  __shared__ float s_x[8][DHEAD];
  __shared__ float s_cf[8][KR];

  int tid = threadIdx.x;

  // static kernel: softmax over taps, per channel
  if (tid < DHEAD) {
    float mx = -1e30f;
#pragma unroll
    for (int j = 0; j < KS; j++) mx = fmaxf(mx, S[j * DHEAD + tid]);
    float sum = 0.f;
#pragma unroll
    for (int j = 0; j < KS; j++) sum += __expf(S[j * DHEAD + tid] - mx);
    float inv = 1.f / sum;
#pragma unroll
    for (int j = 0; j < KS; j++)
      s_kst[j * DHEAD + tid] = __expf(S[j * DHEAD + tid] - mx) * inv;
  }
  __syncthreads();

  int wave  = tid >> 5;
  int lane  = tid & 31;
  int token = blockIdx.x * 8 + wave;
  int n     = token & (SEQ - 1);

  const float* xp = xproj + (size_t)token * D_MODEL + head * DHEAD;
  *(v4f*)(&s_x[wave][lane * 4]) = *(const v4f*)(xp + lane * 4);
  __syncthreads();

  // coeffs[c] = x_row . W[:,c] + b[c]   (W is DHEAD x KR row major)
  for (int c = lane; c < KR; c += 32) {
    float acc = bvec[c];
    for (int d = 0; d < DHEAD; d++) acc += s_x[wave][d] * W[d * KR + c];
    s_cf[wave][c] = acc;
  }
  __syncthreads();

  int d0 = lane * 4;
  v4f V4[RANK];
#pragma unroll
  for (int r = 0; r < RANK; r++) V4[r] = *(const v4f*)(V + r * DHEAD + d0);

  // logits: (coeffs @ V) / sqrt(RANK)
  float lg[KS][4];
#pragma unroll
  for (int j = 0; j < KS; j++) {
    float c0 = s_cf[wave][j * 4 + 0];
    float c1 = s_cf[wave][j * 4 + 1];
    float c2 = s_cf[wave][j * 4 + 2];
    float c3 = s_cf[wave][j * 4 + 3];
#pragma unroll
    for (int i = 0; i < 4; i++)
      lg[j][i] = 0.5f * (c0 * V4[0][i] + c1 * V4[1][i] +
                         c2 * V4[2][i] + c3 * V4[3][i]);
  }

  // softmax over all KS taps (normalization uses every tap, mask comes later)
  float a   = 1.f / (1.f + __expf(-alpha_p[0]));
  float oma = 1.f - a;
  float inv[4];
#pragma unroll
  for (int i = 0; i < 4; i++) {
    float mx = lg[0][i];
#pragma unroll
    for (int j = 1; j < KS; j++) mx = fmaxf(mx, lg[j][i]);
    float sum = 0.f;
#pragma unroll
    for (int j = 0; j < KS; j++) {
      lg[j][i] = __expf(lg[j][i] - mx);
      sum += lg[j][i];
    }
    inv[i] = a / sum;
  }

  // causal masked mix: taps j <= PAD survive; window row = token + j - PAD
  float out[4] = {0.f, 0.f, 0.f, 0.f};
#pragma unroll
  for (int j = 0; j <= PAD; j++) {
    v4f w = {};
    if (n + j - PAD >= 0)
      w = *(const v4f*)(xproj + (size_t)(token + j - PAD) * D_MODEL +
                        head * DHEAD + d0);
#pragma unroll
    for (int i = 0; i < 4; i++) {
      float kh = lg[j][i] * inv[i] + oma * s_kst[j * DHEAD + d0 + i];
      out[i] += kh * w[i];
    }
  }

  v4h hv = {(_Float16)out[0], (_Float16)out[1],
            (_Float16)out[2], (_Float16)out[3]};
  *(v4h*)(mixed + (size_t)token * D_MODEL + head * DHEAD + d0) = hv;
}

// ---------------------------------------------------------------------------
// Launch pipeline
// ---------------------------------------------------------------------------
extern "C" void kernel_launch(void* const* d_in, const int* in_sizes, int n_in,
                              void* d_out, int out_size, void* d_ws,
                              size_t ws_size, hipStream_t stream) {
  const float* x     = (const float*)d_in[0];
  const float* Win   = (const float*)d_in[1];
  const float* b_in  = (const float*)d_in[2];
  const float* Wout  = (const float*)d_in[3];
  const float* b_out = (const float*)d_in[4];
  // d_in[5..12]=gen_W, [13..20]=gen_b, [21..28]=gen_V, [29..36]=gen_S, [37]=alpha
  const float* alpha = (const float*)d_in[37];

  char* ws = (char*)d_ws;
  _Float16* x_h   = (_Float16*)ws;                                  // 32 MB
  _Float16* WinT  = (_Float16*)(ws + (size_t)33554432);             //  2 MB
  _Float16* WoutT = (_Float16*)(ws + (size_t)33554432 + 2097152);   //  2 MB
  float*    xproj = (float*)(ws + (size_t)33554432 + 2 * 2097152);  // 64 MB
  _Float16* mixed =
      (_Float16*)(ws + (size_t)33554432 + 2 * 2097152 + 67108864);  // 32 MB

  // 1) x -> f16
  int n4 = NTOK * D_MODEL / 4;
  cvt_f32_f16<<<(n4 + 255) / 256, 256, 0, stream>>>(x, x_h, n4);

  // 2) transpose + convert weights to (N,K) f16
  dim3 tg(D_MODEL / 256, D_MODEL);
  transpose_cvt<<<tg, 256, 0, stream>>>(Win, WinT, D_MODEL);
  transpose_cvt<<<tg, 256, 0, stream>>>(Wout, WoutT, D_MODEL);

  // 3) x_proj = x @ Win + b_in
  int blocks = (NTOK / 256) * (D_MODEL / 64);  // 64 * 16 = 1024
  gemm_f16_wmma<<<blocks, 256, 0, stream>>>(x_h, WinT, b_in, xproj, NTOK,
                                            D_MODEL, D_MODEL);

  // 4) per-head dynamic kernel mixing -> mixed (f16)
#define LAUNCH_HEAD(h, K)                                                     \
  dka_head<K><<<NTOK / 8, 256, 0, stream>>>(                                  \
      xproj, (const float*)d_in[5 + h], (const float*)d_in[13 + h],           \
      (const float*)d_in[21 + h], (const float*)d_in[29 + h], alpha + h,      \
      mixed, h)
  LAUNCH_HEAD(0, 3);
  LAUNCH_HEAD(1, 3);
  LAUNCH_HEAD(2, 7);
  LAUNCH_HEAD(3, 7);
  LAUNCH_HEAD(4, 11);
  LAUNCH_HEAD(5, 11);
  LAUNCH_HEAD(6, 21);
  LAUNCH_HEAD(7, 21);
#undef LAUNCH_HEAD

  // 5) out = mixed @ Wout + b_out
  gemm_f16_wmma<<<blocks, 256, 0, stream>>>(mixed, WoutT, b_out, (float*)d_out,
                                            NTOK, D_MODEL, D_MODEL);
}
